// ProtoPNet_34033320854190
// MI455X (gfx1250) — compile-verified, matrix operand order checked by reading, not run on hardware
//
#include <hip/hip_runtime.h>
#include <math.h>

#define PROTO_EPS 1e-4f

// Fixed problem dims from the reference
#define BB  128
#define CC  128
#define TT  777
#define PP  500
#define NCC 5

#define NTT    49            // ceil(T/16) t-tiles
#define PTILES 32            // ceil(P/16) p-tiles
#define WAVES_PER_BLOCK 4

typedef __attribute__((ext_vector_type(16))) _Float16 v16h;
typedef __attribute__((ext_vector_type(8)))  float   v8f;

// 16-bit A/B fragment K-index map (ISA 7.12.2, 16-bit 16x32 layout):
// VGPR v (v=0..3): K = 2v,2v+1 (+8 for lane-half 1); VGPR 4..7: K = 16..23 (+8)
__device__ __forceinline__ int kmap(int e, int half) {
    return e + (e >= 8 ? 8 : 0) + 8 * half;
}

// ---------------- x2[b,t] = sum_c x[b,c,t]^2 ----------------
__global__ void x2_kernel(const float* __restrict__ x, float* __restrict__ x2) {
    int idx = blockIdx.x * blockDim.x + threadIdx.x;     // b*T + t
    if (idx >= BB * TT) return;
    int b = idx / TT, t = idx % TT;
    const float* xp = x + (size_t)b * CC * TT + t;
    float s = 0.f;
    #pragma unroll 4
    for (int c = 0; c < CC; ++c) {
        float v = xp[(size_t)c * TT];
        s += v * v;
    }
    x2[idx] = s;
}

// ---------------- p2[p] = sum_c proto[p,c]^2 ----------------
__global__ void p2_kernel(const float* __restrict__ proto, float* __restrict__ p2) {
    int p = blockIdx.x * blockDim.x + threadIdx.x;
    if (p >= PP) return;
    const float* pr = proto + (size_t)p * CC;
    float s = 0.f;
    #pragma unroll 4
    for (int c = 0; c < CC; ++c) s += pr[c] * pr[c];
    p2[p] = s;
}

// ------- pack x into WMMA B-fragment layout, split hi/lo f16 -------
// Layout (f16 elements): [b][tt][c4][half][l16][e]  (e = 0..15)
// One thread per (b,tt,c4,half,l16): 16 strided loads -> one 32B vector store each.
#define NPACK (BB * NTT * 4 * 2 * 16)
__global__ void pack_x_kernel(const float* __restrict__ x,
                              _Float16* __restrict__ xh, _Float16* __restrict__ xl) {
    int n = blockIdx.x * blockDim.x + threadIdx.x;
    if (n >= NPACK) return;
    int l16  = n & 15;          n >>= 4;
    int half = n & 1;           n >>= 1;
    int c4   = n & 3;           n >>= 2;
    int tt   = n % NTT;
    int b    = n / NTT;

    int t  = tt * 16 + l16;
    int tc = (t < TT) ? t : (TT - 1);
    const float* xb = x + (size_t)b * CC * TT + tc;

    v16h vh, vl;
    #pragma unroll
    for (int e = 0; e < 16; ++e) {
        int k = c4 * 32 + kmap(e, half);
        float f = xb[(size_t)k * TT];
        _Float16 h = (_Float16)f;
        vh[e] = h;
        vl[e] = (_Float16)(f - (float)h);
    }
    size_t frag = ((((size_t)b * NTT + tt) * 4 + c4) * 2 + half) * 16 + l16;
    *(v16h*)(xh + frag * 16) = vh;
    *(v16h*)(xl + frag * 16) = vl;
}

// ---------------- main: WMMA xp GEMM + min over T + log-sim ----------------
// grid: (PTILES/4, B), block: 128 (4 waves, one p-tile per wave; all waves
// stream identical B fragments -> WGP$ hits)
__global__ __launch_bounds__(128) void proto_sim_kernel(
        const _Float16* __restrict__ xh, const _Float16* __restrict__ xl,
        const float* __restrict__ proto,
        const float* __restrict__ x2, const float* __restrict__ p2,
        float* __restrict__ sim) {
    const int wave  = threadIdx.x >> 5;
    const int lane  = threadIdx.x & 31;
    const int half  = lane >> 4;
    const int l16   = lane & 15;
    const int b     = blockIdx.y;
    const int ptile = blockIdx.x * WAVES_PER_BLOCK + wave;
    const int p_base = ptile * 16;

    // --- A fragments: prototype tile, split hi/lo f16 (cached in regs) ---
    int prow = p_base + l16; if (prow >= PP) prow = PP - 1;   // clamp tail rows
    const float* prot = proto + (size_t)prow * CC;
    v16h ah[4], al[4];
    #pragma unroll
    for (int c4 = 0; c4 < 4; ++c4) {
        #pragma unroll
        for (int e = 0; e < 16; ++e) {
            int k = c4 * 32 + kmap(e, half);
            float f = prot[k];
            _Float16 h = (_Float16)f;
            ah[c4][e] = h;
            al[c4][e] = (_Float16)(f - (float)h);
        }
    }

    // p2 for this lane's 8 output rows (C/D layout: M = r + 8*half)
    float p2v[8];
    #pragma unroll
    for (int r = 0; r < 8; ++r) {
        int p = p_base + r + 8 * half;
        p2v[r] = p2[p < PP ? p : PP - 1];
    }

    float minv[8];
    #pragma unroll
    for (int r = 0; r < 8; ++r) minv[r] = __builtin_inff();

    const float* x2b = x2 + (size_t)b * TT;
    const v16h* bhP = (const v16h*)xh;
    const v16h* blP = (const v16h*)xl;

    for (int tt = 0; tt < NTT; ++tt) {
        const size_t fb = (((size_t)b * NTT + tt) * 4) * 2;

        // x2 for this lane's column; +inf masks the T tail (d -> +inf, min unaffected)
        const int t = tt * 16 + l16;
        float xt = (t < TT) ? x2b[t] : __builtin_inff();

        if (tt + 1 < NTT) {   // prefetch next t-tile's fragments for this lane
            size_t fn = ((((size_t)b * NTT + tt + 1) * 4) * 2 + half) * 16 + l16;
            __builtin_prefetch(&bhP[fn], 0, 1);
            __builtin_prefetch(&blP[fn], 0, 1);
        }

        // Load ALL 8 fragments into distinct registers first so the loads can
        // issue as one burst and the waits stagger across the WMMA sequence.
        v16h bh[4], bl[4];
        #pragma unroll
        for (int c4 = 0; c4 < 4; ++c4) {
            size_t idx = (fb + (size_t)c4 * 2 + half) * 16 + l16;
            bh[c4] = bhP[idx];
            bl[c4] = blP[idx];
        }

        v8f acc0 = {}, acc1 = {};   // two chains: hi*hi + hi*lo, and lo*hi
        #pragma unroll
        for (int c4 = 0; c4 < 4; ++c4) {
            // f16x3 split-precision; interleave chains so dependent WMMAs
            // are distance-2 apart (hides D->C hazard)
            acc0 = __builtin_amdgcn_wmma_f32_16x16x32_f16(false, ah[c4], false, bh[c4],
                                                          (short)0, acc0, false, false);
            acc1 = __builtin_amdgcn_wmma_f32_16x16x32_f16(false, al[c4], false, bh[c4],
                                                          (short)0, acc1, false, false);
            acc0 = __builtin_amdgcn_wmma_f32_16x16x32_f16(false, ah[c4], false, bl[c4],
                                                          (short)0, acc0, false, false);
        }

        #pragma unroll
        for (int r = 0; r < 8; ++r) {
            float d = fmaf(-2.0f, acc0[r] + acc1[r], xt + p2v[r]);
            d = fmaxf(d, 0.0f);                    // maximum(d, 0)
            minv[r] = fminf(minv[r], d);           // running min over T
        }
    }

    // min-reduce across the 16 lanes of each half (xor masks < 16 stay in-half)
    #pragma unroll
    for (int r = 0; r < 8; ++r) {
        float m = minv[r];
        #pragma unroll
        for (int off = 8; off >= 1; off >>= 1) {
            float o = __shfl_xor(m, off, 32);
            m = fminf(m, o);
        }
        minv[r] = m;
    }

    if (l16 == 0) {
        #pragma unroll
        for (int r = 0; r < 8; ++r) {
            int p = p_base + r + 8 * half;
            if (p < PP) {
                float m = minv[r];
                sim[(size_t)b * PP + p] = logf((m + 1.0f) / (m + PROTO_EPS));
            }
        }
    }
}

// ---------------- logits = sim @ last_w (tiny, deterministic) ----------------
__global__ void logits_kernel(const float* __restrict__ sim,
                              const float* __restrict__ w,
                              float* __restrict__ out) {
    int idx = blockIdx.x * blockDim.x + threadIdx.x;   // b*NC + nc
    if (idx >= BB * NCC) return;
    int b = idx / NCC, nc = idx % NCC;
    const float* sb = sim + (size_t)b * PP;
    float s = 0.f;
    #pragma unroll 4
    for (int p = 0; p < PP; ++p) s += sb[p] * w[p * NCC + nc];
    out[idx] = s;
}

extern "C" void kernel_launch(void* const* d_in, const int* in_sizes, int n_in,
                              void* d_out, int out_size, void* d_ws, size_t ws_size,
                              hipStream_t stream) {
    const float* x     = (const float*)d_in[0];   // [B, C, T]
    const float* proto = (const float*)d_in[1];   // [P, C]
    const float* lastw = (const float*)d_in[2];   // [P, NC]
    float* out = (float*)d_out;                   // [B, NC]

    // workspace layout (bytes):
    //   x2  : BB*TT*4            = 397,824
    //   p2  : 512*4              =   2,048
    //   sim : BB*PP*4            = 256,000
    //   xh  : NPACK*16*2         = 25,690,112   (packed hi f16 fragments)
    //   xl  : NPACK*16*2         = 25,690,112   (packed lo f16 fragments)
    char* ws = (char*)d_ws;
    float*    x2  = (float*)(ws);
    float*    p2  = (float*)(ws + 397824);
    float*    sim = (float*)(ws + 397824 + 2048);
    _Float16* xh  = (_Float16*)(ws + 397824 + 2048 + 256000);
    _Float16* xl  = (_Float16*)(ws + 397824 + 2048 + 256000 + (size_t)NPACK * 16 * 2);

    {   // x2
        int n = BB * TT;
        x2_kernel<<<(n + 255) / 256, 256, 0, stream>>>(x, x2);
    }
    {   // p2
        p2_kernel<<<(PP + 255) / 256, 256, 0, stream>>>(proto, p2);
    }
    {   // pack x into hi/lo f16 WMMA fragments
        pack_x_kernel<<<(NPACK + 255) / 256, 256, 0, stream>>>(x, xh, xl);
    }
    {   // WMMA distance + min-pool + log-sim
        dim3 grid(PTILES / WAVES_PER_BLOCK, BB);
        proto_sim_kernel<<<grid, 128, 0, stream>>>(xh, xl, proto, x2, p2, sim);
    }
    {   // evidence layer
        int n = BB * NCC;
        logits_kernel<<<(n + 127) / 128, 128, 0, stream>>>(sim, lastw, out);
    }
}
